// GaussianBiasedMMD_37855841747666
// MI455X (gfx1250) — matrix operand order repaired and checked by reading, not compiled, not run on hardware
//
#include <hip/hip_runtime.h>
#include <hip/hip_bf16.h>

typedef __attribute__((ext_vector_type(16))) _Float16 v16h;
typedef __attribute__((ext_vector_type(8)))  float    v8f;
typedef __attribute__((__vector_size__(4 * sizeof(int)))) int v4i_vs;

#define SEG 8            // m-dimension split for occupancy
#define BPITCH 68        // LDS row pitch in dwords (64 + 4 pad): conflict-free b128 reads

#define AS1 __attribute__((address_space(1)))
#define AS3 __attribute__((address_space(3)))

// pack two f16 into a dword (element0 = low half)
static __device__ __forceinline__ unsigned packh(_Float16 a, _Float16 b) {
    union { _Float16 h[2]; unsigned u; } x;
    x.h[0] = a; x.h[1] = b;
    return x.u;
}

// load a B fragment (8 dwords = 16 f16) from LDS, 16B-aligned
struct U8x { uint4 a, b; };
static __device__ __forceinline__ v16h ld_bfrag(const unsigned* p) {
    U8x u;
    u.a = *(const uint4*)p;
    u.b = *(const uint4*)(p + 4);
    return __builtin_bit_cast(v16h, u);
}

// ---- CDNA5 async global->LDS copy (ASYNCcnt-tracked), with sync fallback ----
static __device__ __forceinline__ void async_copy_b128(const unsigned* g, unsigned* l) {
#if __has_builtin(__builtin_amdgcn_global_load_async_to_lds_b128)
    __builtin_amdgcn_global_load_async_to_lds_b128(
        (AS1 v4i_vs*)(uintptr_t)g,
        (AS3 v4i_vs*)(unsigned)(uintptr_t)l, 0, 0);
#else
    *(uint4*)l = *(const uint4*)g;
#endif
}

template <int N>
static __device__ __forceinline__ void wait_async() {
#if __has_builtin(__builtin_amdgcn_s_wait_asynccnt)
    __builtin_amdgcn_s_wait_asynccnt(N);
#elif __has_builtin(__builtin_amdgcn_global_load_async_to_lds_b128)
    asm volatile("s_wait_asynccnt %0" :: "i"(N) : "memory");
#endif
}

// build split-f16 A fragments (16x32 layout) + this lane's partial ||x||^2
static __device__ __forceinline__ void build_afrag(const float* __restrict__ Arow, int half,
                                                   v16h ahi[4], v16h alo[4], float& sq) {
    sq = 0.f;
#pragma unroll
    for (int kc = 0; kc < 4; ++kc) {
#pragma unroll
        for (int g = 0; g < 2; ++g) {
            int kb = kc * 32 + half * 16 + g * 8;
#pragma unroll
            for (int j = 0; j < 8; ++j) {
                float x = Arow[kb + j];
                sq += x * x;
                _Float16 h = (_Float16)x;
                _Float16 l = (_Float16)(x - (float)h);
                ahi[kc][g * 8 + j] = h;
                alo[kc][g * 8 + j] = l;
            }
        }
    }
}

// 12 x v_wmma_f32_16x16x32_f16 (3-product split-f16) + exp accumulate
static __device__ __forceinline__ void tile_mma_exp(const unsigned* __restrict__ bHiP,
                                                    const unsigned* __restrict__ bLoP,
                                                    const v16h ahi[4], const v16h alo[4],
                                                    const float x2r[8], float y2c,
                                                    float acc[8]) {
    v8f c0 = {}, c1 = {}, c2 = {};
#pragma unroll
    for (int kc = 0; kc < 4; ++kc) {
        v16h bhi = ld_bfrag(bHiP + kc * 16);
        v16h blo = ld_bfrag(bLoP + kc * 16);
        c0 = __builtin_amdgcn_wmma_f32_16x16x32_f16(false, ahi[kc], false, bhi,
                                                    (short)0, c0, false, false);
        c1 = __builtin_amdgcn_wmma_f32_16x16x32_f16(false, ahi[kc], false, blo,
                                                    (short)0, c1, false, false);
        c2 = __builtin_amdgcn_wmma_f32_16x16x32_f16(false, alo[kc], false, bhi,
                                                    (short)0, c2, false, false);
    }
#pragma unroll
    for (int r = 0; r < 8; ++r) {
        float cross = c0[r] + (c1[r] + c2[r]);
        float d2 = x2r[r] + y2c - 2.0f * cross;
        acc[r] += __expf(-d2);
    }
}

// ---------------------------------------------------------------------------
// Kernel A: y2[j] = ||feat_tar[j]||^2  (exact fp32)
// ---------------------------------------------------------------------------
__global__ void mmd_y2_kernel(const float* __restrict__ ft, float* __restrict__ y2, int m) {
    int j = blockIdx.x * blockDim.x + threadIdx.x;
    if (j >= m) return;
    const float4* p = (const float4*)(ft + (size_t)j * 128);
    float s = 0.f;
#pragma unroll
    for (int i = 0; i < 32; ++i) {
        float4 v = p[i];
        s += v.x * v.x + v.y * v.y + v.z * v.z + v.w * v.w;
    }
    y2[j] = s;
}

// ---------------------------------------------------------------------------
// Kernel P: pack feat_tar into split-f16 hi/lo dword arrays [m][64] each
// ---------------------------------------------------------------------------
__global__ void mmd_pack_kernel(const float* __restrict__ ft,
                                unsigned* __restrict__ hiw, unsigned* __restrict__ low,
                                int ndw) {
    int i = blockIdx.x * blockDim.x + threadIdx.x;   // dword index
    if (i >= ndw) return;
    float2 v = *(const float2*)(ft + 2 * (size_t)i);
    _Float16 h0 = (_Float16)v.x, l0 = (_Float16)(v.x - (float)h0);
    _Float16 h1 = (_Float16)v.y, l1 = (_Float16)(v.y - (float)h1);
    hiw[i] = packh(h0, h1);
    low[i] = packh(l0, l1);
}

// ---------------------------------------------------------------------------
// Kernel B1 (packed path): async double-buffered staging of pre-packed tiles
// ---------------------------------------------------------------------------
__global__ void __launch_bounds__(128)
mmd_partial_packed(const float* __restrict__ feat,
                   const unsigned* __restrict__ hiG, const unsigned* __restrict__ loG,
                   const float* __restrict__ y2, float* __restrict__ partial,
                   int n, int m) {
    __shared__ unsigned ldsT[2][2][16 * BPITCH];   // [buf][hi/lo][row*BPITCH+dw]
    __shared__ float    ldsX2[64];

    const int lane    = threadIdx.x & 31;
    const int wave    = threadIdx.x >> 5;
    const int half    = lane >> 4;
    const int mrow    = lane & 15;
    const int rowBase = blockIdx.x * 64 + wave * 16;
    const int seg     = blockIdx.y;
    const int mseg    = m / SEG;
    const int colBase0 = seg * mseg;
    const int ntiles  = mseg >> 4;

    v16h ahi[4], alo[4];
    float sq;
    build_afrag(feat + (size_t)(rowBase + mrow) * 128, half, ahi, alo, sq);
    float x2full = sq + __shfl_xor(sq, 16);
    if (lane < 16) ldsX2[wave * 16 + lane] = x2full;
    __syncthreads();

    float x2r[8];
#pragma unroll
    for (int r = 0; r < 8; ++r) x2r[r] = ldsX2[wave * 16 + 8 * half + r];
    float acc[8];
#pragma unroll
    for (int r = 0; r < 8; ++r) acc[r] = 0.f;

    const int srow = threadIdx.x >> 3;   // staging: 8 threads per tile row
    const int sseg = threadIdx.x & 7;    // dwords [sseg*8, +8)

    auto issueTile = [&](int tile, int buf) {
        const unsigned* gh = hiG + (((size_t)(colBase0 + tile * 16 + srow)) << 6) + sseg * 8;
        const unsigned* gl = loG + (((size_t)(colBase0 + tile * 16 + srow)) << 6) + sseg * 8;
        unsigned* dh = &ldsT[buf][0][srow * BPITCH + sseg * 8];
        unsigned* dl = &ldsT[buf][1][srow * BPITCH + sseg * 8];
        async_copy_b128(gh,     dh);
        async_copy_b128(gh + 4, dh + 4);
        async_copy_b128(gl,     dl);
        async_copy_b128(gl + 4, dl + 4);
    };

    issueTile(0, 0);
    for (int tile = 0; tile < ntiles; ++tile) {
        const int cur = tile & 1;
        if (tile + 1 < ntiles) {
            issueTile(tile + 1, cur ^ 1);
            wait_async<4>();     // async loads complete in order: current tile resident
        } else {
            wait_async<0>();
        }
        __syncthreads();         // all waves' staging visible

        tile_mma_exp(&ldsT[cur][0][mrow * BPITCH + half * 8],
                     &ldsT[cur][1][mrow * BPITCH + half * 8],
                     ahi, alo, x2r, y2[colBase0 + tile * 16 + mrow], acc);

        __syncthreads();         // buffer safe to overwrite by tile+2 issue
    }

#pragma unroll
    for (int r = 0; r < 8; ++r) {
        acc[r] += __shfl_xor(acc[r], 1);
        acc[r] += __shfl_xor(acc[r], 2);
        acc[r] += __shfl_xor(acc[r], 4);
        acc[r] += __shfl_xor(acc[r], 8);
    }
    if (lane == 0 || lane == 16) {
        int rbase = rowBase + 8 * half;
#pragma unroll
        for (int r = 0; r < 8; ++r)
            partial[(size_t)seg * n + rbase + r] = acc[r];
    }
}

// ---------------------------------------------------------------------------
// Kernel B2 (fallback path): convert tiles in-kernel (round-2 scheme)
// ---------------------------------------------------------------------------
__global__ void __launch_bounds__(128)
mmd_partial_conv(const float* __restrict__ feat, const float* __restrict__ ftar,
                 const float* __restrict__ y2, float* __restrict__ partial,
                 int n, int m) {
    __shared__ unsigned ldsHi[16 * BPITCH];
    __shared__ unsigned ldsLo[16 * BPITCH];
    __shared__ float    ldsX2[64];

    const int lane    = threadIdx.x & 31;
    const int wave    = threadIdx.x >> 5;
    const int half    = lane >> 4;
    const int mrow    = lane & 15;
    const int rowBase = blockIdx.x * 64 + wave * 16;
    const int seg     = blockIdx.y;
    const int mseg    = m / SEG;
    const int colBase0 = seg * mseg;
    const int ntiles  = mseg >> 4;

    v16h ahi[4], alo[4];
    float sq;
    build_afrag(feat + (size_t)(rowBase + mrow) * 128, half, ahi, alo, sq);
    float x2full = sq + __shfl_xor(sq, 16);
    if (lane < 16) ldsX2[wave * 16 + lane] = x2full;
    __syncthreads();

    float x2r[8];
#pragma unroll
    for (int r = 0; r < 8; ++r) x2r[r] = ldsX2[wave * 16 + 8 * half + r];
    float acc[8];
#pragma unroll
    for (int r = 0; r < 8; ++r) acc[r] = 0.f;

    const int srow = threadIdx.x >> 3;
    const int sseg = threadIdx.x & 7;

    for (int tile = 0; tile < ntiles; ++tile) {
        const int colBase = colBase0 + tile * 16;
        __syncthreads();
        {
            const float* src = ftar + (size_t)(colBase + srow) * 128 + sseg * 16;
            unsigned hw[8], lw[8];
#pragma unroll
            for (int q = 0; q < 4; ++q) {
                float4 v = *(const float4*)(src + q * 4);
                _Float16 h0 = (_Float16)v.x, l0 = (_Float16)(v.x - (float)h0);
                _Float16 h1 = (_Float16)v.y, l1 = (_Float16)(v.y - (float)h1);
                _Float16 h2 = (_Float16)v.z, l2 = (_Float16)(v.z - (float)h2);
                _Float16 h3 = (_Float16)v.w, l3 = (_Float16)(v.w - (float)h3);
                hw[2 * q]     = packh(h0, h1);
                hw[2 * q + 1] = packh(h2, h3);
                lw[2 * q]     = packh(l0, l1);
                lw[2 * q + 1] = packh(l2, l3);
            }
            unsigned* dh = &ldsHi[srow * BPITCH + sseg * 8];
            unsigned* dl = &ldsLo[srow * BPITCH + sseg * 8];
            *(uint4*)dh       = make_uint4(hw[0], hw[1], hw[2], hw[3]);
            *(uint4*)(dh + 4) = make_uint4(hw[4], hw[5], hw[6], hw[7]);
            *(uint4*)dl       = make_uint4(lw[0], lw[1], lw[2], lw[3]);
            *(uint4*)(dl + 4) = make_uint4(lw[4], lw[5], lw[6], lw[7]);
        }
        __syncthreads();

        tile_mma_exp(&ldsHi[mrow * BPITCH + half * 8], &ldsLo[mrow * BPITCH + half * 8],
                     ahi, alo, x2r, y2[colBase + mrow], acc);
    }

#pragma unroll
    for (int r = 0; r < 8; ++r) {
        acc[r] += __shfl_xor(acc[r], 1);
        acc[r] += __shfl_xor(acc[r], 2);
        acc[r] += __shfl_xor(acc[r], 4);
        acc[r] += __shfl_xor(acc[r], 8);
    }
    if (lane == 0 || lane == 16) {
        int rbase = rowBase + 8 * half;
#pragma unroll
        for (int r = 0; r < 8; ++r)
            partial[(size_t)seg * n + rbase + r] = acc[r];
    }
}

// ---------------------------------------------------------------------------
// Kernel C: combine segment partials, apply sqrt(k/m^2 - 2k/m + k) + eps
// ---------------------------------------------------------------------------
__global__ void mmd_final_kernel(const float* __restrict__ partial, float* __restrict__ out,
                                 int n, int m) {
    int i = blockIdx.x * blockDim.x + threadIdx.x;
    if (i >= n) return;
    float ks = 0.f;
#pragma unroll
    for (int s = 0; s < SEG; ++s) ks += partial[(size_t)s * n + i];
    float fm = (float)m;
    float v = ks / (fm * fm) - 2.0f * ks / fm + ks;
    out[i] = sqrtf(v) + 1e-16f;
}

// ---------------------------------------------------------------------------
extern "C" void kernel_launch(void* const* d_in, const int* in_sizes, int n_in,
                              void* d_out, int out_size, void* d_ws, size_t ws_size,
                              hipStream_t stream) {
    const float* feat = (const float*)d_in[0];   // [n,128]
    const float* ftar = (const float*)d_in[1];   // [m,128]
    float* out = (float*)d_out;                  // [n]
    const int d = 128;
    const int n = in_sizes[0] / d;               // 8192
    const int m = in_sizes[1] / d;               // 16384

    float* y2      = (float*)d_ws;               // m floats
    float* partial = y2 + m;                     // SEG*n floats

    mmd_y2_kernel<<<(m + 255) / 256, 256, 0, stream>>>(ftar, y2, m);

    dim3 grid(n / 64, SEG);
    const size_t baseFloats  = (size_t)m + (size_t)SEG * n;
    const size_t packedDw    = (size_t)m * 64;                 // per hi / lo array
    const size_t needPacked  = (baseFloats + 2 * packedDw) * 4;

    if (ws_size >= needPacked) {
        unsigned* hiG = (unsigned*)(partial + (size_t)SEG * n);
        unsigned* loG = hiG + packedDw;
        int ndw = m * 64;
        mmd_pack_kernel<<<(ndw + 255) / 256, 256, 0, stream>>>(ftar, hiG, loG, ndw);
        mmd_partial_packed<<<grid, 128, 0, stream>>>(feat, hiG, loG, y2, partial, n, m);
    } else {
        mmd_partial_conv<<<grid, 128, 0, stream>>>(feat, ftar, y2, partial, n, m);
    }

    mmd_final_kernel<<<(n + 255) / 256, 256, 0, stream>>>(partial, out, n, m);
}
